// SwigluMoEBlock_87333864996928
// MI455X (gfx1250) — matrix operand order repaired and checked by reading, not compiled, not run on hardware
//
#include <hip/hip_runtime.h>
#include <hip/hip_bf16.h>

typedef __attribute__((ext_vector_type(16))) __bf16 v16bf;
typedef __attribute__((ext_vector_type(8)))  float  v8f;

#define HDIM   2048
#define IDIM   1024
#define I2DIM  2048
#define NEXP   32
#define TTOT   8192
#define ALPHA  1.702f
#define LIMIT  7.0f

#define ROWS    48                      // tokens per block = 3 WMMA row-tiles
#define RTILES  3
#define THREADS 512                     // 16 waves
#define MAXTIL  ((TTOT + ROWS - 1) / ROWS)   // 171

#define XS_STRIDE 2056                  // 2048 + 8 halfs pad
#define HS_STRIDE 1032                  // 1024 + 8 halfs pad

union FragBF {
    v16bf v;
    unsigned int u[8];
    uint4 q[2];
};

static __device__ __forceinline__ unsigned short f2bf(float f) {
    unsigned int u = __float_as_uint(f);
    unsigned int r = (u + 0x7FFFu + ((u >> 16) & 1u)) >> 16;  // RNE
    return (unsigned short)r;
}
static __device__ __forceinline__ unsigned int pk2(float a, float b) {
    return (unsigned int)f2bf(a) | ((unsigned int)f2bf(b) << 16);
}

// B-fragment loader: 16 contiguous K values for this lane's output column.
template <typename WT>
static __device__ __forceinline__ void loadB(FragBF& b, const WT* bp, int k0) {
    if constexpr (sizeof(WT) == 2) {
        const uint4* q = reinterpret_cast<const uint4*>(bp + k0);
        b.q[0] = q[0];
        b.q[1] = q[1];
    } else {
        const float4* f = reinterpret_cast<const float4*>(bp + k0);
        float4 f0 = f[0], f1 = f[1], f2 = f[2], f3 = f[3];
        b.u[0] = pk2(f0.x, f0.y); b.u[1] = pk2(f0.z, f0.w);
        b.u[2] = pk2(f1.x, f1.y); b.u[3] = pk2(f1.z, f1.w);
        b.u[4] = pk2(f2.x, f2.y); b.u[5] = pk2(f2.z, f2.w);
        b.u[6] = pk2(f3.x, f3.y); b.u[7] = pk2(f3.z, f3.w);
    }
}

// ---------------------------------------------------------------- zero init
__global__ void zero_init(float* __restrict__ out, int n, int* __restrict__ counts) {
    int i = blockIdx.x * blockDim.x + threadIdx.x;
    for (int idx = i; idx < n; idx += gridDim.x * blockDim.x) out[idx] = 0.0f;
    if (i < NEXP) counts[i] = 0;
}

// ---------------------------------------------------------------- fp32 -> bf16
__global__ void convert_bf16(const float* __restrict__ src,
                             unsigned short* __restrict__ dst, int n4) {
    int i = blockIdx.x * blockDim.x + threadIdx.x;
    const float4* s4 = reinterpret_cast<const float4*>(src);
    uint2* d2 = reinterpret_cast<uint2*>(dst);
    for (int idx = i; idx < n4; idx += gridDim.x * blockDim.x) {
        float4 f = s4[idx];
        uint2 o;
        o.x = pk2(f.x, f.y);
        o.y = pk2(f.z, f.w);
        d2[idx] = o;
    }
}

// ---------------------------------------------------------------- router
// One wave per token; lane == expert (E=32 == wave32).
__global__ void router_kernel(const float* __restrict__ x,
                              const float* __restrict__ gw,
                              const float* __restrict__ gb,
                              int* __restrict__ counts,
                              int* __restrict__ toks,
                              float* __restrict__ gates) {
    const int wv   = threadIdx.x >> 5;
    const int lane = threadIdx.x & 31;
    const int tok  = blockIdx.x * 8 + wv;
    if (tok >= TTOT) return;

    const float4* wp = reinterpret_cast<const float4*>(gw + (size_t)lane * HDIM);
    const float4* xp = reinterpret_cast<const float4*>(x + (size_t)tok * HDIM);
    float acc = 0.0f;
    for (int h = 0; h < HDIM / 4; ++h) {
        float4 w4 = wp[h];
        float4 x4 = xp[h];   // same address all lanes -> broadcast
        acc += w4.x * x4.x + w4.y * x4.y + w4.z * x4.z + w4.w * x4.w;
    }
    float logit = acc + gb[lane];

    // softmax over the 32 lanes
    float m = logit;
    for (int off = 16; off; off >>= 1) m = fmaxf(m, __shfl_xor(m, off, 32));
    float p = __expf(logit - m);
    float s = p;
    for (int off = 16; off; off >>= 1) s += __shfl_xor(s, off, 32);
    float prob = p / s;

    // top-1 (tie -> lowest index, matching top_k)
    float v1 = prob; int i1 = lane;
    for (int off = 16; off; off >>= 1) {
        float ov = __shfl_xor(v1, off, 32);
        int   oi = __shfl_xor(i1, off, 32);
        if (ov > v1 || (ov == v1 && oi < i1)) { v1 = ov; i1 = oi; }
    }
    // top-2
    float v2 = (lane == i1) ? -1.0f : prob; int i2 = lane;
    for (int off = 16; off; off >>= 1) {
        float ov = __shfl_xor(v2, off, 32);
        int   oi = __shfl_xor(i2, off, 32);
        if (ov > v2 || (ov == v2 && oi < i2)) { v2 = ov; i2 = oi; }
    }

    if (lane == 0) {
        int p1 = atomicAdd(&counts[i1], 1);
        toks[i1 * TTOT + p1]  = tok;
        gates[i1 * TTOT + p1] = v1;
        int p2 = atomicAdd(&counts[i2], 1);
        toks[i2 * TTOT + p2]  = tok;
        gates[i2 * TTOT + p2] = v2;
    }
}

// ---------------------------------------------------------------- expert FFN
// Block = (expert, tile of 48 tokens); 512 threads = 16 waves (one WGP).
// Each B fragment (expert weights) is reused across 3 row-tiles -> 3x less
// weight traffic. 297 KB LDS tile only fits thanks to CDNA5's 320 KB WGP LDS.
template <typename WT>
__global__ __launch_bounds__(THREADS)
void moe_ffn_kernel(const float* __restrict__ x,
                    const WT*    __restrict__ fc1w,
                    const float* __restrict__ fc1b,
                    const WT*    __restrict__ fc2w,
                    const float* __restrict__ fc2b,
                    const int*   __restrict__ counts,
                    const int*   __restrict__ toks,
                    const float* __restrict__ gates,
                    float* __restrict__ out) {
    __shared__ unsigned short Xs[ROWS * XS_STRIDE];   // token rows, bf16
    __shared__ unsigned short Hs[ROWS * HS_STRIDE];   // swiglu output, bf16
    __shared__ int   sTok[ROWS];
    __shared__ float sGate[ROWS];

    const int e    = blockIdx.x / MAXTIL;   // expert-major for L2 reuse
    const int tile = blockIdx.x % MAXTIL;
    const int cnt  = counts[e];
    const int row0 = tile * ROWS;
    if (row0 >= cnt) return;                // uniform early exit

    const WT*    W1 = fc1w + (size_t)e * I2DIM * HDIM;
    const float* B1 = fc1b + (size_t)e * I2DIM;
    const WT*    W2 = fc2w + (size_t)e * HDIM * IDIM;
    const float* B2 = fc2b + (size_t)e * HDIM;

    const int tid = threadIdx.x;
    if (tid < ROWS) {
        int r = row0 + tid;
        bool ok = (r < cnt);
        sTok[tid]  = ok ? toks[e * TTOT + r]  : -1;
        sGate[tid] = ok ? gates[e * TTOT + r] : 0.0f;
    }
    __syncthreads();

    // ---- Stage A: gather 48 token rows -> LDS (bf16)
    for (int idx = tid; idx < ROWS * HDIM; idx += THREADS) {
        int r = idx >> 11;
        int c = idx & (HDIM - 1);
        int t = sTok[r];
        float v = (t >= 0) ? x[(size_t)t * HDIM + c] : 0.0f;
        Xs[r * XS_STRIDE + c] = f2bf(v);
    }
    __syncthreads();

    const int wv  = tid >> 5;
    const int ln  = tid & 31;
    const int l16 = ln & 15;
    const int hi  = ln >> 4;               // 0: lanes 0-15, 1: lanes 16-31

    // ---- Stage B: fc1 (48 x 2048) = Xs @ W1^T, then SwiGLU -> Hs
    for (int nb = wv * 16; nb < I2DIM; nb += 16 * 16) {
        const int n = nb + l16;
        const WT* bp = W1 + (size_t)n * HDIM + (hi << 4);
        const float bias = B1[n];
        v8f acc[RTILES];
        #pragma unroll
        for (int r = 0; r < RTILES; ++r)
            acc[r] = (v8f){0.f, 0.f, 0.f, 0.f, 0.f, 0.f, 0.f, 0.f};
        #pragma unroll 2
        for (int k0 = 0; k0 < HDIM; k0 += 32) {
            __builtin_prefetch(bp + k0 + 128, 0, 1);
            FragBF b;
            loadB(b, bp, k0);              // one weight fragment ...
            #pragma unroll
            for (int r = 0; r < RTILES; ++r) {  // ... feeds 3 row-tiles
                FragBF a;
                const unsigned short* ap =
                    &Xs[(r * 16 + l16) * XS_STRIDE + k0 + hi * 16];
                a.q[0] = *reinterpret_cast<const uint4*>(ap);
                a.q[1] = *reinterpret_cast<const uint4*>(ap + 8);
                acc[r] = __builtin_amdgcn_wmma_f32_16x16x32_bf16(
                             false, a.v, false, b.v, (short)0, acc[r], false, false);
            }
        }
        // SwiGLU: even (glu) & odd (lin) channels sit in adjacent lanes
        #pragma unroll
        for (int r = 0; r < RTILES; ++r) {
            #pragma unroll
            for (int j = 0; j < 8; ++j) {
                float vv = acc[r][j] + bias;
                float pv = __shfl_xor(vv, 1, 32);
                if ((ln & 1) == 0) {
                    float xg  = fminf(vv, LIMIT);
                    float xl  = fminf(fmaxf(pv, -LIMIT), LIMIT);
                    float sig = __builtin_amdgcn_rcpf(1.0f + __expf(-ALPHA * xg));
                    float rv  = xg * sig * (xl + 1.0f);
                    int m  = r * 16 + j + (hi << 3);
                    int ii = n >> 1;
                    Hs[m * HS_STRIDE + ii] = f2bf(rv);
                }
            }
        }
    }
    __syncthreads();

    // ---- Stage C: fc2 (48 x 2048) = Hs @ W2^T, scatter-accumulate
    for (int nb = wv * 16; nb < HDIM; nb += 16 * 16) {
        const int n = nb + l16;
        const WT* bp = W2 + (size_t)n * IDIM + (hi << 4);
        const float bias = B2[n];
        v8f acc[RTILES];
        #pragma unroll
        for (int r = 0; r < RTILES; ++r)
            acc[r] = (v8f){0.f, 0.f, 0.f, 0.f, 0.f, 0.f, 0.f, 0.f};
        #pragma unroll 2
        for (int k0 = 0; k0 < IDIM; k0 += 32) {
            __builtin_prefetch(bp + k0 + 128, 0, 1);
            FragBF b;
            loadB(b, bp, k0);
            #pragma unroll
            for (int r = 0; r < RTILES; ++r) {
                FragBF a;
                const unsigned short* ap =
                    &Hs[(r * 16 + l16) * HS_STRIDE + k0 + hi * 16];
                a.q[0] = *reinterpret_cast<const uint4*>(ap);
                a.q[1] = *reinterpret_cast<const uint4*>(ap + 8);
                acc[r] = __builtin_amdgcn_wmma_f32_16x16x32_bf16(
                             false, a.v, false, b.v, (short)0, acc[r], false, false);
            }
        }
        #pragma unroll
        for (int r = 0; r < RTILES; ++r) {
            #pragma unroll
            for (int j = 0; j < 8; ++j) {
                int m = r * 16 + j + (hi << 3);
                int t = sTok[m];
                if (t >= 0) {
                    atomicAdd(&out[(size_t)t * HDIM + n],
                              sGate[m] * (acc[r][j] + bias));
                }
            }
        }
    }
}

// ---------------------------------------------------------------- launch
extern "C" void kernel_launch(void* const* d_in, const int* in_sizes, int n_in,
                              void* d_out, int out_size, void* d_ws, size_t ws_size,
                              hipStream_t stream) {
    const float* x   = (const float*)d_in[0];   // [4,2048,2048]
    const float* gw  = (const float*)d_in[1];   // [32,2048]
    const float* gb  = (const float*)d_in[2];   // [32]
    const float* f1w = (const float*)d_in[3];   // [32,2048,2048]
    const float* f1b = (const float*)d_in[4];   // [32,2048]
    const float* f2w = (const float*)d_in[5];   // [32,2048,1024]
    const float* f2b = (const float*)d_in[6];   // [32,2048]
    float* out = (float*)d_out;

    // workspace layout (bytes):
    //   [0,256)      counts | [256,+1MB) token lists | [+1MB) gate lists
    //   [aligned)    fc1_w bf16 (256MB) | fc2_w bf16 (128MB)   (if room)
    char* ws = (char*)d_ws;
    int*   counts = (int*)ws;
    int*   toks   = (int*)(ws + 256);
    float* gates  = (float*)(ws + 256 + (size_t)NEXP * TTOT * 4);
    size_t base   = 256 + 2ull * NEXP * TTOT * 4;
    base = (base + 255) & ~255ull;
    unsigned short* w1b = (unsigned short*)(ws + base);
    const size_t n1 = (size_t)NEXP * I2DIM * HDIM;          // 134M elems
    const size_t n2 = (size_t)NEXP * HDIM * IDIM;           //  67M elems
    unsigned short* w2b = w1b + n1;
    const size_t need = base + 2 * (n1 + n2);

    zero_init<<<4096, 256, 0, stream>>>(out, out_size, counts);
    router_kernel<<<TTOT / 8, 256, 0, stream>>>(x, gw, gb, counts, toks, gates);

    if (ws_size >= need) {
        // Pre-convert weights to bf16: hot loop is pure b128 loads + v_wmma.
        convert_bf16<<<8192, 256, 0, stream>>>(f1w, w1b, (int)(n1 / 4));
        convert_bf16<<<8192, 256, 0, stream>>>(f2w, w2b, (int)(n2 / 4));
        moe_ffn_kernel<unsigned short><<<NEXP * MAXTIL, THREADS, 0, stream>>>(
            x, w1b, f1b, w2b, f2b, counts, toks, gates, out);
    } else {
        // Fallback: convert weights inline in the K-loop.
        moe_ffn_kernel<float><<<NEXP * MAXTIL, THREADS, 0, stream>>>(
            x, f1w, f1b, f2w, f2b, counts, toks, gates, out);
    }
}
